// LinearPQSTE_49890340110827
// MI455X (gfx1250) — matrix operand order repaired and checked by reading, not compiled, not run on hardware
//
#include <hip/hip_runtime.h>

// LinearPQSTE forward for MI455X (gfx1250, wave32, WMMA + async-to-LDS).
//
//   K2: W f32 [o][d] -> Wbf bf16 [o][d]   (pure convert; [o][d] is already the
//       K-contiguous orientation the WMMA B-fragment wants)
//   K1: PQ quantize via bf16 WMMA distance GEMM + shfl_xor argmin -> x_hat bf16
//   K3: out = x_hat @ Wbf^T, 128x128 block tiles, double-buffered LDS staging
//       via GLOBAL_LOAD_ASYNC_TO_LDS_B128 (fallback: sync uint4 copies)
//
// Workspace: x_hat 64 MB + Wbf 2 MB = 66 MB in d_ws.

#define N_TOKENS 32768
#define D_IN     1024
#define D_OUT    1024
#define N_SUB    16
#define N_CODE   256
#define SUB_D    64

typedef __attribute__((ext_vector_type(16))) __bf16 v16bf;
typedef __attribute__((ext_vector_type(8)))  float  v8f;
typedef int v4i_vs __attribute__((vector_size(16)));   // matches builtin param sugar

union Frag {
  v16bf v;
  uint4 q[2];
  unsigned short h[16];
};

#ifndef __has_builtin
#define __has_builtin(x) 0
#endif
#if __has_builtin(__builtin_amdgcn_global_load_async_to_lds_b128)
#define ASYNC_LDS 1
#else
#define ASYNC_LDS 0
#endif

__device__ __forceinline__ void copy_b128(const unsigned short* g, unsigned short* l) {
#if ASYNC_LDS
  __builtin_amdgcn_global_load_async_to_lds_b128(
      (__attribute__((address_space(1))) v4i_vs*)g,
      (__attribute__((address_space(3))) v4i_vs*)l, 0, 0);
#else
  *(uint4*)l = *(const uint4*)g;
#endif
}

__device__ __forceinline__ void wait_async() {
#if ASYNC_LDS
#if __has_builtin(__builtin_amdgcn_s_wait_asynccnt)
  __builtin_amdgcn_s_wait_asynccnt(0);
#else
  asm volatile("s_wait_asynccnt 0" ::: "memory");
#endif
#endif
}

__device__ __forceinline__ unsigned short f2bf(float f) {
  unsigned u = __float_as_uint(f);
  u += 0x7FFFu + ((u >> 16) & 1u);              // round-to-nearest-even
  return (unsigned short)(u >> 16);
}
__device__ __forceinline__ unsigned pack2bf(float a, float b) {
  unsigned ua = __float_as_uint(a);
  unsigned ub = __float_as_uint(b);
  ua += 0x7FFFu + ((ua >> 16) & 1u);
  ub += 0x7FFFu + ((ub >> 16) & 1u);
  return (ua >> 16) | (ub & 0xFFFF0000u);
}

// ---------------------------------------------------------------------------
// Kernel 2: W[o][d] f32 -> Wbf[o][d] bf16 (coalesced convert, no transpose)
// ---------------------------------------------------------------------------
__global__ __launch_bounds__(256)
void weight_to_bf16(const float* __restrict__ wsrc, unsigned short* __restrict__ wbf) {
  int e8 = blockIdx.x * 256 + threadIdx.x;      // 8 floats per thread
  const float4* src = (const float4*)(wsrc) + e8 * 2;
  float4 f0 = src[0], f1 = src[1];
  uint4 o;
  o.x = pack2bf(f0.x, f0.y);
  o.y = pack2bf(f0.z, f0.w);
  o.z = pack2bf(f1.x, f1.y);
  o.w = pack2bf(f1.z, f1.w);
  ((uint4*)wbf)[e8] = o;
}

// ---------------------------------------------------------------------------
// Kernel 1: PQ quantize. Block = 256 thr (8 waves) = 128 tokens x 1 subspace.
// dots[token, code] via WMMA (M=16/wave, N=256 as 16 column tiles, K=64).
// score = ||c||^2 - 2*dot; argmin per token; gather codeword -> x_hat (bf16).
// ---------------------------------------------------------------------------
#define Q_TOK    128
#define LDSTRIDE 72   // 64 + 8 pad elements; 144 B row, 16B-aligned chunks

__global__ __launch_bounds__(256)
void pq_quantize(const float* __restrict__ x, const float* __restrict__ cbooks,
                 unsigned short* __restrict__ xhat) {
  __shared__ unsigned short xLds[Q_TOK * LDSTRIDE];   // 18.0 KB
  __shared__ unsigned short cbLds[N_CODE * LDSTRIDE]; // 36.0 KB
  __shared__ float csq[N_CODE];                       //  1.0 KB

  const int t   = threadIdx.x;
  const int s   = blockIdx.y;                 // subspace
  const int gm0 = blockIdx.x * Q_TOK;         // first token of block

  // --- stage x subvectors (f32 -> bf16): token r = t>>1, 32-dim segment ---
  {
    int r = t >> 1, seg = (t & 1) * 32;
    const float4* src = (const float4*)(x + (size_t)(gm0 + r) * D_IN + s * SUB_D + seg);
    unsigned* dst = (unsigned*)xLds + r * (LDSTRIDE / 2) + seg / 2;
#pragma unroll
    for (int i = 0; i < 8; ++i) {
      float4 f = src[i];
      dst[i * 2 + 0] = pack2bf(f.x, f.y);
      dst[i * 2 + 1] = pack2bf(f.z, f.w);
    }
  }
  // --- stage codebook row t (one codeword per thread) + ||c||^2 ---
  {
    const float4* src = (const float4*)(cbooks + ((size_t)s * N_CODE + t) * SUB_D);
    unsigned* dst = (unsigned*)cbLds + t * (LDSTRIDE / 2);
    float sq = 0.f;
#pragma unroll
    for (int i = 0; i < 16; ++i) {
      float4 f = src[i];
      sq += f.x * f.x + f.y * f.y + f.z * f.z + f.w * f.w;
      dst[i * 2 + 0] = pack2bf(f.x, f.y);
      dst[i * 2 + 1] = pack2bf(f.z, f.w);
    }
    csq[t] = sq;
  }
  __syncthreads();

  const int w = t >> 5, lane = t & 31, li = lane & 15, hi = lane >> 4;

  v8f acc[16];
#pragma unroll
  for (int c = 0; c < 16; ++c)
#pragma unroll
    for (int v = 0; v < 8; ++v) acc[c][v] = 0.f;

  // K = 64 dims in 2 steps of 32; 16 column tiles of 16 codes each.
#pragma unroll
  for (int ks = 0; ks < 2; ++ks) {
    Frag a;
    // A layout (16-bit 16x32): lane li = row; K runs {8*hi..8*hi+7, 16+8*hi..}
    const unsigned short* ap = xLds + (w * 16 + li) * LDSTRIDE + ks * 32 + 8 * hi;
    a.q[0] = *(const uint4*)(ap);
    a.q[1] = *(const uint4*)(ap + 16);
#pragma unroll
    for (int ct = 0; ct < 16; ++ct) {
      Frag b;
      // B layout (16-bit 32x16): lane holds column N=code; K runs {8*hi..8*hi+7, 16+8*hi..}
      const unsigned short* bp = cbLds + (ct * 16 + li) * LDSTRIDE + ks * 32 + 8 * hi;
      b.q[0] = *(const uint4*)(bp);
      b.q[1] = *(const uint4*)(bp + 16);
      acc[ct] = __builtin_amdgcn_wmma_f32_16x16x32_bf16(
          false, a.v, false, b.v, (short)0, acc[ct], false, false);
    }
  }

  // C layout: lane holds column N = li; VGPR v -> row M = v + 8*hi.
#pragma unroll
  for (int v = 0; v < 8; ++v) {
    float minv = 3.4e38f;
    int   mini = 0;
#pragma unroll
    for (int ct = 0; ct < 16; ++ct) {
      int k = ct * 16 + li;
      float sc = csq[k] - 2.0f * acc[ct][v];
      if (sc < minv) { minv = sc; mini = k; }
    }
#pragma unroll
    for (int m = 1; m < 16; m <<= 1) {
      float ov = __shfl_xor(minv, m, 32);
      int   oi = __shfl_xor(mini, m, 32);
      if (ov < minv || (ov == minv && oi < mini)) { minv = ov; mini = oi; }
    }
    // all 16 lanes of the half agree on mini; copy codeword (64 bf16 = 32 dwords)
    int tok = gm0 + w * 16 + v + 8 * hi;
    const unsigned* srcu = (const unsigned*)cbLds + mini * (LDSTRIDE / 2);
    unsigned* dstu = (unsigned*)xhat + (size_t)tok * (D_IN / 2) + s * (SUB_D / 2);
    dstu[li]      = srcu[li];
    dstu[li + 16] = srcu[li + 16];
  }
}

// ---------------------------------------------------------------------------
// Kernel 3: out[N, D_OUT] = x_hat(bf16) @ Wbf^T.
// Block tile 128x128, 8 waves (4x2), each wave 32x64 (2x4 v8f accums), BK=32.
// Double-buffered LDS staging via async-to-LDS b128 copies.
// ---------------------------------------------------------------------------
#define G_BM 128
#define G_BN 128
#define G_BK 32
#define ASTR 40   // 32 + 8 pad; 80 B rows keep 16 B chunk alignment
#define KT   (D_IN / G_BK)

__global__ __launch_bounds__(256)
void gemm_xhat_w(const unsigned short* __restrict__ xhat,
                 const unsigned short* __restrict__ wbf,
                 float* __restrict__ out) {
  __shared__ unsigned short Atile[2][G_BM * ASTR];  // 2 x 10 KB  [token][k]
  __shared__ unsigned short Btile[2][G_BN * ASTR];  // 2 x 10 KB  [ocol][k]

  const int t   = threadIdx.x;
  const int gm0 = blockIdx.x * G_BM;
  const int gn0 = blockIdx.y * G_BN;
  const int w = t >> 5, lane = t & 31, li = lane & 15, hi = lane >> 4;
  const int wmi = w >> 1, wnj = w & 1;            // 4 x 2 wave grid, 32x64 each

  v8f acc[2][4];
#pragma unroll
  for (int i = 0; i < 2; ++i)
#pragma unroll
    for (int j = 0; j < 4; ++j)
#pragma unroll
      for (int v = 0; v < 8; ++v) acc[i][j][v] = 0.f;

  // Stage one K-tile (A: 128x32, B: 128x32 bf16) into buffer `buf`.
  // 512 + 512 b128 chunks; each wave issues 2+2 instructions (1 chunk/lane).
  auto stage = [&](int buf, int k0) {
#pragma unroll
    for (int p = 0; p < 2; ++p) {
      int id  = p * 256 + w * 32 + lane;          // 0..511
      int row = id >> 2, c = (id & 3) * 8;
      copy_b128(xhat + (size_t)(gm0 + row) * D_IN + k0 + c,
                &Atile[buf][row * ASTR + c]);
    }
#pragma unroll
    for (int p = 0; p < 2; ++p) {
      int id  = p * 256 + w * 32 + lane;
      int row = id >> 2, c = (id & 3) * 8;
      copy_b128(wbf + (size_t)(gn0 + row) * D_IN + k0 + c,
                &Btile[buf][row * ASTR + c]);
    }
  };

  stage(0, 0);
  for (int kt = 0; kt < KT; ++kt) {
    const int cur = kt & 1;
    wait_async();
    __syncthreads();                 // buf[cur] full; buf[cur^1] fully consumed
    if (kt + 1 < KT) stage(cur ^ 1, (kt + 1) * G_BK);

    Frag a[2], b[4];
#pragma unroll
    for (int i = 0; i < 2; ++i) {
      const unsigned short* ap = &Atile[cur][(wmi * 32 + i * 16 + li) * ASTR + 8 * hi];
      a[i].q[0] = *(const uint4*)(ap);
      a[i].q[1] = *(const uint4*)(ap + 16);
    }
#pragma unroll
    for (int j = 0; j < 4; ++j) {
      const unsigned short* bp = &Btile[cur][(wnj * 64 + j * 16 + li) * ASTR + 8 * hi];
      b[j].q[0] = *(const uint4*)(bp);
      b[j].q[1] = *(const uint4*)(bp + 16);
    }
#pragma unroll
    for (int i = 0; i < 2; ++i)
#pragma unroll
      for (int j = 0; j < 4; ++j)
        acc[i][j] = __builtin_amdgcn_wmma_f32_16x16x32_bf16(
            false, a[i].v, false, b[j].v, (short)0, acc[i][j], false, false);
    __syncthreads();                 // done reading buf[cur] before it refills
  }

  // C layout: col = li (+tile offset); VGPR v -> row v + 8*hi.
#pragma unroll
  for (int i = 0; i < 2; ++i)
#pragma unroll
    for (int j = 0; j < 4; ++j) {
      int col   = gn0 + wnj * 64 + j * 16 + li;
      int rbase = gm0 + wmi * 32 + i * 16 + 8 * hi;
#pragma unroll
      for (int v = 0; v < 8; ++v)
        out[(size_t)(rbase + v) * D_OUT + col] = acc[i][j][v];
    }
}

// ---------------------------------------------------------------------------
extern "C" void kernel_launch(void* const* d_in, const int* in_sizes, int n_in,
                              void* d_out, int out_size, void* d_ws, size_t ws_size,
                              hipStream_t stream) {
  (void)in_sizes; (void)n_in; (void)out_size; (void)ws_size;
  const float* x  = (const float*)d_in[0];   // [32768, 1024]
  const float* wg = (const float*)d_in[1];   // [1024, 1024]
  const float* cb = (const float*)d_in[2];   // [16, 256, 64]
  float* out = (float*)d_out;                // [32768, 1024] f32

  unsigned short* xhat = (unsigned short*)d_ws;                  // 64 MB bf16
  unsigned short* wbf  = xhat + (size_t)N_TOKENS * D_IN;         //  2 MB bf16

  hipLaunchKernelGGL(weight_to_bf16, dim3((D_OUT * D_IN / 8) / 256), dim3(256),
                     0, stream, wg, wbf);
  hipLaunchKernelGGL(pq_quantize, dim3(N_TOKENS / Q_TOK, N_SUB), dim3(256),
                     0, stream, x, cb, xhat);
  hipLaunchKernelGGL(gemm_xhat_w, dim3(N_TOKENS / G_BM, D_OUT / G_BN), dim3(256),
                     0, stream, xhat, wbf, out);
}
